// Attention_79061757985231
// MI455X (gfx1250) — compile-verified, matrix operand order checked by reading, not run on hardware
//
#include <hip/hip_runtime.h>
#include <math.h>

// ---------- types ----------
typedef __attribute__((ext_vector_type(16))) __bf16          v16bf;
typedef __attribute__((ext_vector_type(8)))  float           v8f;
typedef __attribute__((ext_vector_type(8)))  unsigned short  v8us;
typedef __attribute__((ext_vector_type(4)))  unsigned int    v4u;
typedef __attribute__((ext_vector_type(8)))  unsigned int    v8u;
typedef __attribute__((ext_vector_type(4)))  int             v4i;
typedef __attribute__((ext_vector_type(8)))  int             v8i;

// f32 -> bf16 round-to-nearest-even (result in low 16 bits)
__device__ __forceinline__ unsigned f2bf(float f) {
    unsigned u = __builtin_bit_cast(unsigned, f);
    unsigned rb = 0x7FFFu + ((u >> 16) & 1u);
    return (u + rb) >> 16;
}

// =====================================================================
// Tensor Data Mover: DMA a tile_rows x 32-elem bf16 tile (row-major,
// row stride = tensor_k elems) into LDS. Descriptor pad feature inserts
// 16B padding after every 64B row -> 80B LDS pitch (bank-conflict free,
// 16B-aligned rows for ds_load_b128 fragment reads).
// =====================================================================
__device__ __forceinline__ void tdm_load_tile_2d(unsigned lds_byte_addr,
                                                 const void* global_tile,
                                                 unsigned tensor_k,
                                                 unsigned tensor_rows,
                                                 unsigned tile_rows)
{
    const unsigned long long ga = (unsigned long long)(uintptr_t)global_tile;
    v4u g0;
    g0[0] = 1u;                                                  // count=1, user descriptor
    g0[1] = lds_byte_addr;                                       // lds_addr
    g0[2] = (unsigned)ga;                                        // global_addr[31:0]
    g0[3] = (unsigned)((ga >> 32) & 0x01FFFFFFu) | (2u << 30);   // addr[56:32] | type=2
    v8u g1;
    g1[0] = (1u << 16)        // data_size = 2 bytes
          | (1u << 20)        // pad_enable
          | (3u << 22)        // pad_interval: 2^(3+1)=16 dwords (= one 64B tile row)
          | (3u << 25);       // pad_amount: 4 dwords (16B)
    g1[1] = (tensor_k & 0xFFFFu) << 16;                              // tensor_dim0[15:0]
    g1[2] = ((tensor_k >> 16) & 0xFFFFu) | ((tensor_rows & 0xFFFFu) << 16);
    g1[3] = ((tensor_rows >> 16) & 0xFFFFu) | (32u << 16);           // tile_dim0 = 32
    g1[4] = tile_rows & 0xFFFFu;                                     // tile_dim1, tile_dim2 = 0
    g1[5] = tensor_k;                                                // tensor_dim0_stride[31:0]
    g1[6] = 0u;
    g1[7] = 0u;
    const v4i z4 = {0, 0, 0, 0};
#if defined(__clang_major__) && (__clang_major__ >= 23)
    const v8i z8 = {0, 0, 0, 0, 0, 0, 0, 0};
    __builtin_amdgcn_tensor_load_to_lds(g0, __builtin_bit_cast(v8i, g1), z4, z4, z8, 0);
#else
    __builtin_amdgcn_tensor_load_to_lds(g0, __builtin_bit_cast(v8i, g1), z4, z4, 0);
#endif
}

// =====================================================================
// bf16-native WMMA GEMM with TDM double-buffered staging.
//   C[M,N](f32) = A[M,K](bf16) * B[N,K](bf16)^T  (both K-contiguous)
//   Block tile 256x128, 8 waves as 4(m) x 2(n), wave tile 64x64:
//   16 WMMAs per K-step for 16 ds_load_b128 (1.0 LDS load / WMMA).
//   CAUSAL: epilogue keeps (col<=row)*scale else -inf
//   Cbf (optional): row-major bf16 copy of C for downstream GEMMs
// =====================================================================
constexpr int A_SH    = 256 * 40;            // shorts per padded A tile (80B x 256 rows)
constexpr int B_SH    = 128 * 40;            // shorts per padded B tile
constexpr int BUF_SH  = A_SH + B_SH;         // 15360 shorts per buffer

template <bool CAUSAL>
__global__ __launch_bounds__(256)
void gemm_tdm_bf16(const unsigned short* __restrict__ A,
                   const unsigned short* __restrict__ B,
                   float* __restrict__ C,
                   unsigned short* __restrict__ Cbf,
                   int M, int N, int K, float scale)
{
    __shared__ __align__(16) unsigned short lds[2 * BUF_SH];   // 60 KB
    const unsigned ldsBase = (unsigned)(uintptr_t)(&lds[0]);   // low 32b == LDS offset

    const int tid  = threadIdx.x;
    const int lane = tid & 31;
    const int wave = tid >> 5;
    const int wm   = wave >> 1;        // 0..3 : 64-row slab
    const int wn   = wave & 1;         // 0..1 : 64-col slab
    const int half = lane >> 4;
    const int l16  = lane & 15;
    const int kfr  = half * 8;

    const int mBase = blockIdx.y * 256;
    const int nBase = blockIdx.x * 128;
    const int nK    = K >> 5;

    auto issue = [&](int kt, int buf) {
        const unsigned base = ldsBase + (unsigned)buf * (BUF_SH * 2u);
        tdm_load_tile_2d(base,             A + (size_t)mBase * K + (size_t)kt * 32,
                         (unsigned)K, (unsigned)M, 256u);
        tdm_load_tile_2d(base + A_SH * 2u, B + (size_t)nBase * K + (size_t)kt * 32,
                         (unsigned)K, (unsigned)N, 128u);
    };

    const v8f vzero = {0.f,0.f,0.f,0.f,0.f,0.f,0.f,0.f};
    v8f acc[4][4];
#pragma unroll
    for (int i = 0; i < 4; ++i)
#pragma unroll
        for (int j = 0; j < 4; ++j) acc[i][j] = vzero;

    if (tid < 32) issue(0, 0);          // wave 0 drives the TDM pipeline

    for (int kt = 0; kt < nK; ++kt) {
        const int buf = kt & 1;
        if (tid < 32) __builtin_amdgcn_s_wait_tensorcnt(0);
        __syncthreads();                               // tile visible to all waves
        if (kt + 1 < nK && tid < 32) issue(kt + 1, buf ^ 1);

        const unsigned short* aT = &lds[buf * BUF_SH];
        const unsigned short* bT = aT + A_SH;

        v16bf aF[4], bF[4];
#pragma unroll
        for (int tm = 0; tm < 4; ++tm) {
            const unsigned short* p = aT + (wm * 64 + tm * 16 + l16) * 40;
            union { v8us h[2]; v16bf f; } u;
            u.h[0] = *reinterpret_cast<const v8us*>(p + kfr);
            u.h[1] = *reinterpret_cast<const v8us*>(p + kfr + 16);
            aF[tm] = u.f;
        }
#pragma unroll
        for (int tn = 0; tn < 4; ++tn) {
            const unsigned short* p = bT + (wn * 64 + tn * 16 + l16) * 40;
            union { v8us h[2]; v16bf f; } u;
            u.h[0] = *reinterpret_cast<const v8us*>(p + kfr);
            u.h[1] = *reinterpret_cast<const v8us*>(p + kfr + 16);
            bF[tn] = u.f;
        }

#pragma unroll
        for (int tm = 0; tm < 4; ++tm)
#pragma unroll
            for (int tn = 0; tn < 4; ++tn)
                acc[tm][tn] = __builtin_amdgcn_wmma_f32_16x16x32_bf16(
                    false, aF[tm], false, bF[tn], (short)0, acc[tm][tn], false, false);

        __syncthreads();                               // readers done before buf reuse
    }

#pragma unroll
    for (int tm = 0; tm < 4; ++tm) {
#pragma unroll
        for (int tn = 0; tn < 4; ++tn) {
            const int rowB = mBase + wm * 64 + tm * 16 + half * 8;
            const int col  = nBase + wn * 64 + tn * 16 + l16;
#pragma unroll
            for (int r = 0; r < 8; ++r) {
                const int row = rowB + r;
                float v = acc[tm][tn][r] * scale;
                if (CAUSAL) v = (col <= row) ? v : -__builtin_inff();
                C[(size_t)row * N + col] = v;
                if (Cbf) Cbf[(size_t)row * N + col] = (unsigned short)f2bf(v);
            }
        }
    }
}

// ---------- one-shot f32 -> bf16 convert (vectorized) ----------
__global__ __launch_bounds__(256)
void convert_f32_bf16(const float* __restrict__ in, unsigned short* __restrict__ out, size_t n4)
{
    const size_t i = (size_t)blockIdx.x * 256 + threadIdx.x;
    if (i >= n4) return;
    const float4 v = reinterpret_cast<const float4*>(in)[i];
    uint2 pk;
    pk.x = f2bf(v.x) | (f2bf(v.y) << 16);
    pk.y = f2bf(v.z) | (f2bf(v.w) << 16);
    reinterpret_cast<uint2*>(out)[i] = pk;
}

// ---------- tiled transpose + convert: in[R][C] f32 -> out[C][R] bf16 ----------
__global__ __launch_bounds__(256)
void transpose_f32_bf16(const float* __restrict__ in, unsigned short* __restrict__ out,
                        int R, int Cc)
{
    __shared__ float t[32][33];
    const int c0 = blockIdx.x * 32, r0 = blockIdx.y * 32;
    const int tx = threadIdx.x, ty = threadIdx.y;   // block (32, 8)
#pragma unroll
    for (int i = 0; i < 32; i += 8)
        t[ty + i][tx] = in[(size_t)(r0 + ty + i) * Cc + (c0 + tx)];
    __syncthreads();
#pragma unroll
    for (int i = 0; i < 32; i += 8)
        out[(size_t)(c0 + ty + i) * R + (r0 + tx)] = (unsigned short)f2bf(t[tx][ty + i]);
}

// ---------- row softmax (in place) + optional bf16 copy ----------
__global__ __launch_bounds__(256)
void softmax_rows(float* __restrict__ s, int n, unsigned short* __restrict__ obf)
{
    float* p = s + (size_t)blockIdx.x * n;
    __shared__ float red[256];
    const int tid = threadIdx.x;

    float m = -__builtin_inff();
    for (int c = tid; c < n; c += 256) m = fmaxf(m, p[c]);
    red[tid] = m; __syncthreads();
    for (int off = 128; off > 0; off >>= 1) {
        if (tid < off) red[tid] = fmaxf(red[tid], red[tid + off]);
        __syncthreads();
    }
    m = red[0]; __syncthreads();

    float sum = 0.f;
    for (int c = tid; c < n; c += 256) {
        const float e = __expf(p[c] - m);
        p[c] = e;
        sum += e;
    }
    red[tid] = sum; __syncthreads();
    for (int off = 128; off > 0; off >>= 1) {
        if (tid < off) red[tid] += red[tid + off];
        __syncthreads();
    }
    const float inv = 1.f / red[0];
    for (int c = tid; c < n; c += 256) {
        const float v = p[c] * inv;
        p[c] = v;
        if (obf) obf[(size_t)blockIdx.x * n + c] = (unsigned short)f2bf(v);
    }
}

// =====================================================================
// Fallback GEMM (round-1 style): f32 operands, convert while staging.
// Used only when the workspace is too small for bf16 operand copies.
// =====================================================================
constexpr int BM = 128, BN = 128, BK = 32;
constexpr int LDS_STRIDE = BK + 8;

template <bool TRANS_B, bool CAUSAL>
__global__ __launch_bounds__(256)
void gemm_bf16_wmma(const float* __restrict__ A, const float* __restrict__ B,
                    float* __restrict__ C, int M, int N, int K, float scale)
{
    __shared__ __align__(16) unsigned short aS[BM * LDS_STRIDE];
    __shared__ __align__(16) unsigned short bS[BN * LDS_STRIDE];

    const int tid  = threadIdx.x;
    const int lane = tid & 31;
    const int wave = tid >> 5;
    const int wm   = wave >> 2;
    const int wn   = wave & 3;
    const int half = lane >> 4;
    const int l16  = lane & 15;
    const int kfr  = half * 8;

    const int mBase = blockIdx.y * BM;
    const int nBase = blockIdx.x * BN;

    const v8f vzero = {0.f,0.f,0.f,0.f,0.f,0.f,0.f,0.f};
    v8f acc[4][2];
#pragma unroll
    for (int i = 0; i < 4; ++i)
#pragma unroll
        for (int j = 0; j < 2; ++j) acc[i][j] = vzero;

    const int nK = K >> 5;
    for (int kt = 0; kt < nK; ++kt) {
        const int k0 = kt * BK;
#pragma unroll
        for (int c = tid; c < BM * BK / 4; c += 256) {
            const int r  = c >> 3;
            const int kc = (c & 7) << 2;
            const float4 v = *reinterpret_cast<const float4*>(A + (size_t)(mBase + r) * K + k0 + kc);
            uint2 pk;
            pk.x = f2bf(v.x) | (f2bf(v.y) << 16);
            pk.y = f2bf(v.z) | (f2bf(v.w) << 16);
            *reinterpret_cast<uint2*>(&aS[r * LDS_STRIDE + kc]) = pk;
        }
        if (TRANS_B) {
#pragma unroll
            for (int c = tid; c < BN * BK / 4; c += 256) {
                const int r  = c >> 3;
                const int kc = (c & 7) << 2;
                const float4 v = *reinterpret_cast<const float4*>(B + (size_t)(nBase + r) * K + k0 + kc);
                uint2 pk;
                pk.x = f2bf(v.x) | (f2bf(v.y) << 16);
                pk.y = f2bf(v.z) | (f2bf(v.w) << 16);
                *reinterpret_cast<uint2*>(&bS[r * LDS_STRIDE + kc]) = pk;
            }
        } else {
#pragma unroll
            for (int c = tid; c < BK * BN / 4; c += 256) {
                const int kr = c >> 5;
                const int nc = (c & 31) << 2;
                const float4 v = *reinterpret_cast<const float4*>(B + (size_t)(k0 + kr) * N + nBase + nc);
                bS[(nc + 0) * LDS_STRIDE + kr] = (unsigned short)f2bf(v.x);
                bS[(nc + 1) * LDS_STRIDE + kr] = (unsigned short)f2bf(v.y);
                bS[(nc + 2) * LDS_STRIDE + kr] = (unsigned short)f2bf(v.z);
                bS[(nc + 3) * LDS_STRIDE + kr] = (unsigned short)f2bf(v.w);
            }
        }
        __syncthreads();

        v16bf aF[4], bF[2];
#pragma unroll
        for (int tm = 0; tm < 4; ++tm) {
            const unsigned short* p = &aS[(wm * 64 + tm * 16 + l16) * LDS_STRIDE];
            union { v8us h[2]; v16bf f; } u;
            u.h[0] = *reinterpret_cast<const v8us*>(p + kfr);
            u.h[1] = *reinterpret_cast<const v8us*>(p + kfr + 16);
            aF[tm] = u.f;
        }
#pragma unroll
        for (int tn = 0; tn < 2; ++tn) {
            const unsigned short* p = &bS[(wn * 32 + tn * 16 + l16) * LDS_STRIDE];
            union { v8us h[2]; v16bf f; } u;
            u.h[0] = *reinterpret_cast<const v8us*>(p + kfr);
            u.h[1] = *reinterpret_cast<const v8us*>(p + kfr + 16);
            bF[tn] = u.f;
        }
#pragma unroll
        for (int tm = 0; tm < 4; ++tm)
#pragma unroll
            for (int tn = 0; tn < 2; ++tn)
                acc[tm][tn] = __builtin_amdgcn_wmma_f32_16x16x32_bf16(
                    false, aF[tm], false, bF[tn], (short)0, acc[tm][tn], false, false);
        __syncthreads();
    }

#pragma unroll
    for (int tm = 0; tm < 4; ++tm) {
#pragma unroll
        for (int tn = 0; tn < 2; ++tn) {
            const int rowB = mBase + wm * 64 + tm * 16 + half * 8;
            const int col  = nBase + wn * 32 + tn * 16 + l16;
#pragma unroll
            for (int r = 0; r < 8; ++r) {
                const int row = rowB + r;
                float v = acc[tm][tn][r] * scale;
                if (CAUSAL) v = (col <= row) ? v : -__builtin_inff();
                C[(size_t)row * N + col] = v;
            }
        }
    }
}

// ---------- host launch ----------
extern "C" void kernel_launch(void* const* d_in, const int* in_sizes, int n_in,
                              void* d_out, int out_size, void* d_ws, size_t ws_size,
                              hipStream_t stream)
{
    (void)in_sizes; (void)n_in; (void)out_size;

    constexpr int    S = 4096, D = 2048, E = 2048;
    constexpr size_t MEG = 1u << 20;

    const float* x  = (const float*)d_in[0];
    const float* wq = (const float*)d_in[1];
    const float* wk = (const float*)d_in[2];
    const float* wv = (const float*)d_in[3];

    float* out = (float*)d_out;
    float* z   = out;
    float* sc  = z  + (size_t)S * E;
    float* xq  = sc + (size_t)S * S;
    float* xv  = xq + (size_t)S * E;
    float* xk  = xv + (size_t)S * E;

    const float invdk = 1.0f / sqrtf((float)E);
    const dim3  blk(256);

    const size_t NEED = 60 * MEG * 2;   // 120 MB of bf16 operand copies
    if (ws_size >= NEED && d_ws) {
        unsigned short* w     = (unsigned short*)d_ws;
        unsigned short* xbf   = w;             // [S][D]   8M
        unsigned short* wqT   = w + 8  * MEG;  // [E][D]   4M
        unsigned short* wkT   = w + 12 * MEG;
        unsigned short* wvT   = w + 16 * MEG;
        unsigned short* xqbf  = w + 20 * MEG;  // [S][E]   8M
        unsigned short* xkbf  = w + 28 * MEG;  // [S][E]
        unsigned short* xvT   = w + 36 * MEG;  // [E][S]   8M
        unsigned short* scbf  = w + 44 * MEG;  // [S][S]  16M

        // one-shot operand conversion (x direct; weights transposed to [N][K])
        convert_f32_bf16<<<dim3((unsigned)(((size_t)S * D / 4 + 255) / 256)), blk, 0, stream>>>(
            x, xbf, (size_t)S * D / 4);
        transpose_f32_bf16<<<dim3(E / 32, D / 32), dim3(32, 8), 0, stream>>>(wq, wqT, D, E);
        transpose_f32_bf16<<<dim3(E / 32, D / 32), dim3(32, 8), 0, stream>>>(wk, wkT, D, E);
        transpose_f32_bf16<<<dim3(E / 32, D / 32), dim3(32, 8), 0, stream>>>(wv, wvT, D, E);

        const dim3 gQKV(E / 128, S / 256);   // 16 x 16
        const dim3 gSS (S / 128, S / 256);   // 32 x 16

        // QKV projections (TDM-staged, bf16 operands)
        gemm_tdm_bf16<false><<<gQKV, blk, 0, stream>>>(xbf, wqT, xq, xqbf, S, E, D, 1.0f);
        gemm_tdm_bf16<false><<<gQKV, blk, 0, stream>>>(xbf, wkT, xk, xkbf, S, E, D, 1.0f);
        gemm_tdm_bf16<false><<<gQKV, blk, 0, stream>>>(xbf, wvT, xv, nullptr, S, E, D, 1.0f);
        transpose_f32_bf16<<<dim3(E / 32, S / 32), dim3(32, 8), 0, stream>>>(xv, xvT, S, E);

        // masked scaled logits, softmax, PV
        gemm_tdm_bf16<true><<<gSS, blk, 0, stream>>>(xqbf, xkbf, sc, nullptr, S, S, E, invdk);
        softmax_rows<<<dim3(S), blk, 0, stream>>>(sc, S, scbf);
        gemm_tdm_bf16<false><<<gQKV, blk, 0, stream>>>(scbf, xvT, z, nullptr, S, E, S, 1.0f);
    } else {
        // fallback: convert-in-kernel f32 path
        const dim3 gQKV(E / 128, S / 128);
        const dim3 gSS (S / 128, S / 128);
        gemm_bf16_wmma<false, false><<<gQKV, blk, 0, stream>>>(x, wq, xq, S, E, D, 1.0f);
        gemm_bf16_wmma<false, false><<<gQKV, blk, 0, stream>>>(x, wk, xk, S, E, D, 1.0f);
        gemm_bf16_wmma<false, false><<<gQKV, blk, 0, stream>>>(x, wv, xv, S, E, D, 1.0f);
        gemm_bf16_wmma<true,  true ><<<gSS,  blk, 0, stream>>>(xq, xk, sc, S, S, E, invdk);
        softmax_rows<<<dim3(S), blk, 0, stream>>>(sc, S, nullptr);
        gemm_bf16_wmma<false, false><<<gQKV, blk, 0, stream>>>(sc, xv, z, S, E, S, 1.0f);
    }
}
